// CELPNetSub_47012712022158
// MI455X (gfx1250) — matrix operand order repaired and checked by reading, not compiled, not run on hardware
//
#include <hip/hip_runtime.h>
#include <math.h>

#define BATCH 65536
#define SUB   40
#define COND  256

typedef __bf16 bf16;
typedef __attribute__((ext_vector_type(16))) __bf16 v16bf;
typedef __attribute__((ext_vector_type(8)))  float  v8f;

// ---------------- WMMA fragment helpers (wave32, 16x16x32 bf16) -------------

// A-matrix 16x32 (MxK), 16-bit: lane<16 holds row M=lane, K = {k0..k0+7, k0+16..k0+23}
//                               lane>=16 holds row M=lane-16, K = {k0+8..15, k0+24..31}
__device__ __forceinline__ v16bf load_a_frag(const bf16* X, int ld, int m0, int k0) {
    int lane = threadIdx.x & 31;
    int lr = lane & 15, hf = lane >> 4;
    const bf16* p = X + (size_t)(m0 + lr) * ld + k0 + hf * 8;
    union { uint4 u[2]; v16bf v; } un;
    un.u[0] = *reinterpret_cast<const uint4*>(p);        // K +0..7
    un.u[1] = *reinterpret_cast<const uint4*>(p + 16);   // K +16..23
    return un.v;
}

// B-matrix 32x16 (KxN) fragment from an LDS-staged, row-padded weight slice.
// lane<16 = column rbase+lane, K = {k0..k0+7, k0+16..k0+23} ; lane>=16 same column, K = {k0+8..15, k0+24..31}.
__device__ __forceinline__ v16bf load_b_lds(const bf16* sW, int stride, int rbase, int k0) {
    int lane = threadIdx.x & 31;
    int lr = lane & 15, hf = lane >> 4;
    const bf16* p = sW + (rbase + lr) * stride + k0 + hf * 8;
    union { uint4 u[2]; v16bf v; } un;
    un.u[0] = *reinterpret_cast<const uint4*>(p);        // K +0..7
    un.u[1] = *reinterpret_cast<const uint4*>(p + 16);   // K +16..23
    return un.v;
}

__device__ __forceinline__ v8f wmma_bf16(v16bf a, v16bf b, v8f c) {
    return __builtin_amdgcn_wmma_f32_16x16x32_bf16(false, a, false, b, (short)0, c,
                                                   false, false);
}

__device__ __forceinline__ v8f vzero8() {
    v8f z = {0.f, 0.f, 0.f, 0.f, 0.f, 0.f, 0.f, 0.f};
    return z;
}

__device__ __forceinline__ float sigmoidf_(float x) { return 1.0f / (1.0f + expf(-x)); }

// ---------------- utility kernels -------------------------------------------

__global__ void pad_w_kernel(const float* __restrict__ src, bf16* __restrict__ dst,
                             int N, int K, int Npad, int Kpad) {
    int idx = blockIdx.x * blockDim.x + threadIdx.x;
    if (idx >= Npad * Kpad) return;
    int n = idx / Kpad, k = idx - n * Kpad;
    float v = (n < N && k < K) ? src[n * K + k] : 0.0f;
    dst[idx] = (bf16)v;
}

__global__ void set_gain_row_kernel(const float* __restrict__ gain_w, bf16* __restrict__ Wout) {
    int k = blockIdx.x * blockDim.x + threadIdx.x;
    if (k < COND) Wout[40 * COND + k] = (bf16)gain_w[k];
}

__global__ void f2bf_kernel(const float* __restrict__ src, bf16* __restrict__ dst, int n) {
    int i = blockIdx.x * blockDim.x + threadIdx.x;
    int stride = gridDim.x * blockDim.x;
    for (; i < n; i += stride) dst[i] = (bf16)src[i];
}

// Build X0 bf16 [B,384]: [cond(256) | prev/(1e-5+||prev||)(40) | log(1e-5+||prev||)(1)
//                         | phase(80) | zeros(7)]   -- one wave per row
__global__ void prep_kernel(const float* __restrict__ cond, const float* __restrict__ prev,
                            const float* __restrict__ phase, bf16* __restrict__ X0) {
    int row = blockIdx.x;
    int lane = threadIdx.x;
    const float* pr = prev + (size_t)row * SUB;
    float v0 = pr[lane];
    float s = v0 * v0;
    if (lane < 8) { float v1 = pr[32 + lane]; s += v1 * v1; }
#pragma unroll
    for (int o = 16; o > 0; o >>= 1) s += __shfl_xor(s, o, 32);
    float gain = sqrtf(s);
    float inv  = 1.0f / (1e-5f + gain);
    float lg   = logf(1e-5f + gain);

    bf16* xr = X0 + (size_t)row * 384;
    const float* cd = cond  + (size_t)row * COND;
    const float* ph = phase + (size_t)row * (2 * SUB);
#pragma unroll
    for (int j = 0; j < 12; ++j) {
        int c = lane + 32 * j;
        float v;
        if (c < 256)      v = cd[c];
        else if (c < 296) v = pr[c - 256] * inv;
        else if (c == 296) v = lg;
        else if (c < 377) v = ph[c - 297];
        else              v = 0.0f;
        xr[c] = (bf16)v;
    }
}

// ---------------- dense + tanh (Y = tanh(X W^T + b)), bf16 out --------------
// 16 waves / block share a 128-column weight slice in LDS (row stride Kpad+8,
// conflict-free); each wave owns a 16-row M-tile and 8 N-tiles (128 cols).
// B-fragments are batch-loaded into distinct registers so the ds_loads clause
// together and drain with graduated s_wait_dscnt while WMMAs overlap.
// grid = (B/256, N/128), block = 512, LDS = 128*(Kpad+8)*2 bytes.
__global__ void dense_tanh_kernel(const bf16* __restrict__ X, int ldx,
                                  const bf16* __restrict__ W, int Kpad,
                                  const float* __restrict__ bias,
                                  bf16* __restrict__ Y, int ldy) {
    extern __shared__ bf16 smem[];
    const int stride = Kpad + 8;
    const int chunks = Kpad >> 3;               // uint4 per weight row
    int tid = threadIdx.x, wave = tid >> 5;
    int m0 = (blockIdx.x * 16 + wave) * 16;
    int n0 = blockIdx.y * 128;

    for (int idx = tid; idx < 128 * chunks; idx += 512) {
        int rs = idx / chunks, c = idx - rs * chunks;
        reinterpret_cast<uint4*>(smem + rs * stride)[c] =
            reinterpret_cast<const uint4*>(W + (size_t)(n0 + rs) * Kpad)[c];
    }
    __syncthreads();

    v8f acc[8];
#pragma unroll
    for (int t = 0; t < 8; ++t) acc[t] = vzero8();
    for (int k = 0; k < Kpad; k += 32) {
        v16bf a = load_a_frag(X, ldx, m0, k);
        v16bf bfr[8];
#pragma unroll
        for (int t = 0; t < 8; ++t) bfr[t] = load_b_lds(smem, stride, t * 16, k);
#pragma unroll
        for (int t = 0; t < 8; ++t) acc[t] = wmma_bf16(a, bfr[t], acc[t]);
    }
    int lane = tid & 31;
    int lr = lane & 15, hf = lane >> 4;
#pragma unroll
    for (int t = 0; t < 8; ++t) {
        int col = n0 + t * 16 + lr;
        float bb = bias[col];
#pragma unroll
        for (int r = 0; r < 8; ++r) {
            int row = m0 + r + 8 * hf;
            Y[(size_t)row * ldy + col] = (bf16)tanhf(acc[t][r] + bb);
        }
    }
}

// ---------------- fused GRU cell --------------------------------------------
// gi = x@Wih^T + bih ; gh = h@Whh^T + bhh   (Wih/Whh: [768,256], gates r,z,n)
// g  = (1-z)*tanh(i_n + r*h_n) + z*h
// 16 waves / block; the 32-column slice of Wih and Whh for all 3 gates
// (2 * 96 rows * 264 bf16 = 101376 B) is staged in LDS and shared by all waves.
// Each wave: 16 rows x 32 cols -> 12 f32 accumulators (96 VGPRs). The 6 B-
// fragments per input matrix are batch-loaded before their WMMA group.
// grid = (B/256, 256/32), block = 512.
#define GRU_STR 264          // 256 + 8 pad  -> conflict-free ds_load_b128
__global__ void gru_fused_kernel(const bf16* __restrict__ Xb,  // [B,256] bf16 x
                                 const bf16* __restrict__ Hb,  // [B,256] bf16 h
                                 const float* __restrict__ Hf, // [B,256] f32 h
                                 const bf16* __restrict__ Wih, // [768,256] bf16
                                 const bf16* __restrict__ Whh, // [768,256] bf16
                                 const float* __restrict__ bih,
                                 const float* __restrict__ bhh,
                                 float* __restrict__ Gout,     // [B,256] f32
                                 bf16* __restrict__ Gbf) {     // [B,256] bf16
    extern __shared__ bf16 smem[];
    bf16* sWih = smem;                  // 96 rows (3 gates x 32 cols) x GRU_STR
    bf16* sWhh = smem + 96 * GRU_STR;
    int tid = threadIdx.x, wave = tid >> 5;
    int m0 = (blockIdx.x * 16 + wave) * 16;
    int n0 = blockIdx.y * 32;

    for (int idx = tid; idx < 96 * 32; idx += 512) {     // 32 uint4 chunks / row
        int rs = idx >> 5, c = idx & 31;
        int gr = (rs >> 5) * COND + n0 + (rs & 31);      // global weight row
        reinterpret_cast<uint4*>(sWih + rs * GRU_STR)[c] =
            reinterpret_cast<const uint4*>(Wih + (size_t)gr * COND)[c];
        reinterpret_cast<uint4*>(sWhh + rs * GRU_STR)[c] =
            reinterpret_cast<const uint4*>(Whh + (size_t)gr * COND)[c];
    }
    __syncthreads();

    v8f acc_i[6], acc_h[6];              // [gate*2 + tile]
#pragma unroll
    for (int q = 0; q < 6; ++q) { acc_i[q] = vzero8(); acc_h[q] = vzero8(); }

    for (int k = 0; k < COND; k += 32) {
        v16bf ax = load_a_frag(Xb, COND, m0, k);
        v16bf ah = load_a_frag(Hb, COND, m0, k);
        v16bf bb[6];
#pragma unroll
        for (int q = 0; q < 6; ++q) bb[q] = load_b_lds(sWih, GRU_STR, q * 16, k);
#pragma unroll
        for (int q = 0; q < 6; ++q) acc_i[q] = wmma_bf16(ax, bb[q], acc_i[q]);
#pragma unroll
        for (int q = 0; q < 6; ++q) bb[q] = load_b_lds(sWhh, GRU_STR, q * 16, k);
#pragma unroll
        for (int q = 0; q < 6; ++q) acc_h[q] = wmma_bf16(ah, bb[q], acc_h[q]);
    }

    int lane = tid & 31;
    int lr = lane & 15, hf = lane >> 4;
#pragma unroll
    for (int t = 0; t < 2; ++t) {
        int col = n0 + t * 16 + lr;
        float bir = bih[col],            bhr = bhh[col];
        float biz = bih[COND + col],     bhz = bhh[COND + col];
        float bin = bih[2 * COND + col], bhn = bhh[2 * COND + col];
#pragma unroll
        for (int r = 0; r < 8; ++r) {
            int row = m0 + r + 8 * hf;
            float ir = acc_i[0 + t][r] + bir, hr = acc_h[0 + t][r] + bhr;
            float iz = acc_i[2 + t][r] + biz, hz = acc_h[2 + t][r] + bhz;
            float in_ = acc_i[4 + t][r] + bin, hn = acc_h[4 + t][r] + bhn;
            float rg = sigmoidf_(ir + hr);
            float zg = sigmoidf_(iz + hz);
            float ng = tanhf(in_ + rg * hn);
            float hv = Hf[(size_t)row * COND + col];
            float gv = (1.0f - zg) * ng + zg * hv;
            Gout[(size_t)row * COND + col] = gv;
            Gbf [(size_t)row * COND + col] = (bf16)gv;
        }
    }
}

// ---------------- output head -----------------------------------------------
// Wout bf16 [48,256]: rows 0..39 = dout_w, row 40 = gain_w, rows 41..47 = 0.
// sig[row, c<40] = tanh(acc_c + dout_b[c]) * exp(acc_40 + gain_b)
// 16 waves / block; full Wout staged into LDS. grid = B/256, block = 512.
__global__ void head_kernel(const bf16* __restrict__ G3, const bf16* __restrict__ Wout,
                            const float* __restrict__ dout_b, const float* __restrict__ gain_b,
                            float* __restrict__ Sig) {
    extern __shared__ bf16 smem[];
    int tid = threadIdx.x, wave = tid >> 5;
    int m0 = (blockIdx.x * 16 + wave) * 16;

    for (int idx = tid; idx < 48 * 32; idx += 512) {
        int rs = idx >> 5, c = idx & 31;
        reinterpret_cast<uint4*>(smem + rs * GRU_STR)[c] =
            reinterpret_cast<const uint4*>(Wout + (size_t)rs * COND)[c];
    }
    __syncthreads();

    v8f acc[3];
#pragma unroll
    for (int t = 0; t < 3; ++t) acc[t] = vzero8();
    for (int k = 0; k < COND; k += 32) {
        v16bf a = load_a_frag(G3, COND, m0, k);
        v16bf bfr[3];
#pragma unroll
        for (int t = 0; t < 3; ++t) bfr[t] = load_b_lds(smem, GRU_STR, t * 16, k);
#pragma unroll
        for (int t = 0; t < 3; ++t) acc[t] = wmma_bf16(a, bfr[t], acc[t]);
    }
    int lane = tid & 31;
    int lr = lane & 15, hf = lane >> 4;
    float gb = gain_b[0];
    float db0 = dout_b[lr];
    float db1 = dout_b[16 + lr];
    float db2 = (lr < 8) ? dout_b[32 + lr] : 0.0f;
    int gsrc = 8 + (hf << 4);   // lane holding col 40 for this half's rows
#pragma unroll
    for (int r = 0; r < 8; ++r) {
        float egain = expf(__shfl(acc[2][r], gsrc, 32) + gb);
        int row = m0 + r + 8 * hf;
        Sig[(size_t)row * SUB + lr]      = tanhf(acc[0][r] + db0) * egain;
        Sig[(size_t)row * SUB + 16 + lr] = tanhf(acc[1][r] + db1) * egain;
        if (lr < 8)
            Sig[(size_t)row * SUB + 32 + lr] = tanhf(acc[2][r] + db2) * egain;
    }
}

// ---------------- launcher ---------------------------------------------------

extern "C" void kernel_launch(void* const* d_in, const int* in_sizes, int n_in,
                              void* d_out, int out_size, void* d_ws, size_t ws_size,
                              hipStream_t stream) {
    (void)in_sizes; (void)n_in; (void)out_size; (void)ws_size;
    const size_t B = BATCH;

    const float* cond   = (const float*)d_in[0];
    const float* prev   = (const float*)d_in[1];
    const float* phase  = (const float*)d_in[2];
    const float* h1     = (const float*)d_in[3];
    const float* h2     = (const float*)d_in[4];
    const float* h3     = (const float*)d_in[5];
    const float* d1_w   = (const float*)d_in[6];
    const float* d1_b   = (const float*)d_in[7];
    const float* d2_w   = (const float*)d_in[8];
    const float* d2_b   = (const float*)d_in[9];
    const float* dout_w = (const float*)d_in[10];
    const float* dout_b = (const float*)d_in[11];
    const float* gain_w = (const float*)d_in[12];
    const float* gain_b = (const float*)d_in[13];
    const float* w_ih[3] = { (const float*)d_in[14], (const float*)d_in[18], (const float*)d_in[22] };
    const float* w_hh[3] = { (const float*)d_in[15], (const float*)d_in[19], (const float*)d_in[23] };
    const float* b_ih[3] = { (const float*)d_in[16], (const float*)d_in[20], (const float*)d_in[24] };
    const float* b_hh[3] = { (const float*)d_in[17], (const float*)d_in[21], (const float*)d_in[25] };

    float* out  = (float*)d_out;
    float* sig  = out;                       // [B,40]
    float* g1o  = out + B * SUB;             // [B,256]
    float* g2o  = g1o + B * COND;
    float* g3o  = g2o + B * COND;

    // workspace layout (bf16 buffers), all 16B aligned
    char* ws = (char*)d_ws;
    bf16* X0  = (bf16*)(ws);                              size_t o = B * 384 * 2;
    bf16* Xa  = (bf16*)(ws + o);                          o += B * COND * 2;
    bf16* Xb  = (bf16*)(ws + o);                          o += B * COND * 2;
    bf16* Hb  = (bf16*)(ws + o);                          o += B * COND * 2;
    bf16* Wd1 = (bf16*)(ws + o);                          o += 256 * 384 * 2;
    bf16* Wd2 = (bf16*)(ws + o);                          o += 256 * 256 * 2;
    bf16* Wih[3], *Whh[3];
    for (int i = 0; i < 3; ++i) { Wih[i] = (bf16*)(ws + o); o += 768 * 256 * 2; }
    for (int i = 0; i < 3; ++i) { Whh[i] = (bf16*)(ws + o); o += 768 * 256 * 2; }
    bf16* Wout = (bf16*)(ws + o);                         o += 48 * 256 * 2;

    // ---- weight conversion / packing ----
    auto padgrid = [](int n) { return (n + 255) / 256; };
    pad_w_kernel<<<padgrid(256 * 384), 256, 0, stream>>>(d1_w, Wd1, 256, 377, 256, 384);
    pad_w_kernel<<<padgrid(256 * 256), 256, 0, stream>>>(d2_w, Wd2, 256, 256, 256, 256);
    for (int i = 0; i < 3; ++i) {
        pad_w_kernel<<<padgrid(768 * 256), 256, 0, stream>>>(w_ih[i], Wih[i], 768, 256, 768, 256);
        pad_w_kernel<<<padgrid(768 * 256), 256, 0, stream>>>(w_hh[i], Whh[i], 768, 256, 768, 256);
    }
    pad_w_kernel<<<padgrid(48 * 256), 256, 0, stream>>>(dout_w, Wout, 40, 256, 48, 256);
    set_gain_row_kernel<<<1, 256, 0, stream>>>(gain_w, Wout);

    // ---- input feature build ----
    prep_kernel<<<(int)B, 32, 0, stream>>>(cond, prev, phase, X0);

    // ---- dense stack (16 waves / block, 128-col LDS weight slice) ----
    dim3 gD((int)(B / 256), COND / 128);
    dense_tanh_kernel<<<gD, 512, (size_t)128 * (384 + 8) * 2, stream>>>(X0, 384, Wd1, 384, d1_b, Xa, COND);
    dense_tanh_kernel<<<gD, 512, (size_t)128 * (256 + 8) * 2, stream>>>(Xa, COND, Wd2, 256, d2_b, Xb, COND);

    // ---- GRU stack (h -> bf16, then fused GEMM+gating, LDS-staged weights) ----
    int nH = (int)(B * COND);
    dim3 gG((int)(B / 256), COND / 32);
    size_t gruLds = (size_t)2 * 96 * GRU_STR * 2;   // 101376 B
    f2bf_kernel<<<4096, 256, 0, stream>>>(h1, Hb, nH);
    gru_fused_kernel<<<gG, 512, gruLds, stream>>>(Xb, Hb, h1, Wih[0], Whh[0], b_ih[0], b_hh[0], g1o, Xa);
    f2bf_kernel<<<4096, 256, 0, stream>>>(h2, Hb, nH);
    gru_fused_kernel<<<gG, 512, gruLds, stream>>>(Xa, Hb, h2, Wih[1], Whh[1], b_ih[1], b_hh[1], g2o, Xb);
    f2bf_kernel<<<4096, 256, 0, stream>>>(h3, Hb, nH);
    gru_fused_kernel<<<gG, 512, gruLds, stream>>>(Xb, Hb, h3, Wih[2], Whh[2], b_ih[2], b_hh[2], g3o, Xa);

    // ---- output head ----
    head_kernel<<<(int)(B / 256), 512, (size_t)48 * GRU_STR * 2, stream>>>(Xa, Wout, dout_b, gain_b, sig);
}
